// LlmcViTSelfAttention_5841155522792
// MI455X (gfx1250) — matrix operand order repaired and checked by reading, not compiled, not run on hardware
//
#include <hip/hip_runtime.h>

typedef _Float16 f16;
typedef __attribute__((ext_vector_type(16))) _Float16 v16h;
typedef __attribute__((ext_vector_type(8)))  _Float16 v8h;
typedef __attribute__((ext_vector_type(8)))  float    v8f;

#define BB 32
#define SS 577
#define DD 1024
#define HH 16
#define DH 64
#define SP 608                  // padded key length: 19 * 32
#define MROWS (BB * SS)         // 18464
#define MT32  (MROWS / 32)      // 577 tiles of 32 rows (exact)

// ---------- fragment helpers ----------

// f16 A-fragment (16x32, M x K). M = lane%16,
// half h: K = (lane/16)*16 + h  -> one contiguous 16-half group (matches B).
__device__ __forceinline__ v16h load_a16(const f16* p, int laneHi) {
  return *(const v16h*)(p + laneHi * 16);
}

__device__ __forceinline__ v8f wmma_f16(v16h a, v16h b, v8f c) {
  return __builtin_amdgcn_wmma_f32_16x16x32_f16(false, a, false, b, (short)0, c,
                                                false, false);
}

// ---------- conversion kernels ----------

__global__ void f32_to_f16_kernel(const float* __restrict__ src,
                                  f16* __restrict__ dst, int n) {
  int i = blockIdx.x * blockDim.x + threadIdx.x;
  int stride = gridDim.x * blockDim.x;
  for (; i < n; i += stride) dst[i] = (f16)src[i];
}

// Wt[n*1024 + k] = (f16) W[k*1024 + n]  (coalesced writes)
__global__ void w_transpose_f16_kernel(const float* __restrict__ w,
                                       f16* __restrict__ wt) {
  int i = blockIdx.x * blockDim.x + threadIdx.x;
  int stride = gridDim.x * blockDim.x;
  for (; i < DD * DD; i += stride) {
    int n = i >> 10, k = i & 1023;
    wt[i] = (f16)w[(size_t)k * DD + n];
  }
}

// ---------- QKV projection GEMM ----------
// Each wave computes a 32(M) x 64(N) tile: 2 A-fragments share every
// B-fragment (1.5 vmem per WMMA).  out = X*W + bias, scattered to
// attention layouts.
// mode 0: Q -> [bh, SP, 64]; mode 1: K -> [bh, SP, 64]; mode 2: V^T -> [bh, 64, SP]
__global__ void __launch_bounds__(256)
qkv_gemm_kernel(const f16* __restrict__ xh, const f16* __restrict__ wt,
                const float* __restrict__ bias, f16* __restrict__ out, int mode) {
  int lane = threadIdx.x & 31;
  int laneLo = lane & 15, laneHi = lane >> 4;
  int mtile = blockIdx.x * 8 + (threadIdx.x >> 5);   // 32-row tiles
  if (mtile >= MT32) return;
  int nbase = blockIdx.y * 64;

  const f16* arow0 = xh + (size_t)(mtile * 32 + laneLo) * DD;
  const f16* arow1 = arow0 + (size_t)16 * DD;

  v8f acc[2][4];
#pragma unroll
  for (int mt = 0; mt < 2; ++mt)
#pragma unroll
    for (int j = 0; j < 4; ++j) acc[mt][j] = {};

  for (int kk = 0; kk < DD; kk += 32) {
    // pull the A stream ahead of the WMMA pipe (global_prefetch_b8)
    if (kk + 128 < DD) {
      __builtin_prefetch(arow0 + kk + 128, 0, 3);
      __builtin_prefetch(arow1 + kk + 128, 0, 3);
    }
    v16h a0 = load_a16(arow0 + kk, laneHi);
    v16h a1 = load_a16(arow1 + kk, laneHi);
#pragma unroll
    for (int j = 0; j < 4; ++j) {
      const f16* bp =
          wt + (size_t)(nbase + j * 16 + laneLo) * DD + kk + laneHi * 16;
      v16h b = *(const v16h*)bp;
      acc[0][j] = wmma_f16(a0, b, acc[0][j]);
      acc[1][j] = wmma_f16(a1, b, acc[1][j]);
    }
  }

#pragma unroll
  for (int j = 0; j < 4; ++j) {
    int n = nbase + j * 16 + laneLo;
    float bv = bias[n];
    int hh = n >> 6, d = n & 63;
#pragma unroll
    for (int mt = 0; mt < 2; ++mt) {
#pragma unroll
      for (int r = 0; r < 8; ++r) {
        int m = mtile * 32 + mt * 16 + r + 8 * laneHi;  // flat row = b*577 + s
        int b = m / SS;
        int s = m - b * SS;
        int bh = b * HH + hh;
        f16 c = (f16)(acc[mt][j][r] + bv);
        if (mode == 2) out[((size_t)bh * DH + d) * SP + s] = c;  // V transposed
        else           out[((size_t)bh * SP + s) * DH + d] = c;  // Q / K rows
      }
    }
  }
}

// ---------- flash attention (one wave per (bh, 16-query tile)) ----------

__global__ void __launch_bounds__(32)
attn_kernel(const f16* __restrict__ qws, const f16* __restrict__ kws,
            const f16* __restrict__ vtws, float* __restrict__ out) {
  int lane = threadIdx.x & 31;
  int laneLo = lane & 15, laneHi = lane >> 4;
  int bh = blockIdx.x;
  int b = bh >> 4, h = bh & 15;
  int qt = blockIdx.y;

  __shared__ __align__(32) f16 lds_p[16 * 32];

  const f16* qrow = qws + ((size_t)bh * SP + qt * 16 + laneLo) * DH;
  v16h qa0 = load_a16(qrow, laneHi);        // dh 0..31
  v16h qa1 = load_a16(qrow + 32, laneHi);   // dh 32..63

  v8f acc[4];
#pragma unroll
  for (int j = 0; j < 4; ++j) acc[j] = {};
  float mrow[8], lrow[8];
#pragma unroll
  for (int r = 0; r < 8; ++r) { mrow[r] = -1e30f; lrow[r] = 0.0f; }

  for (int kc = 0; kc < SP; kc += 32) {
#pragma unroll
    for (int kt = 0; kt < 2; ++kt) {
      int key = kc + kt * 16 + laneLo;
      const f16* kp = kws + ((size_t)bh * SP + key) * DH + laneHi * 16;
      v16h kb0 = *(const v16h*)kp;          // dh 0..31 slice
      v16h kb1 = *(const v16h*)(kp + 32);   // dh 32..63 slice
      v8f sf = {};
      sf = wmma_f16(qa0, kb0, sf);
      sf = wmma_f16(qa1, kb1, sf);
      bool maskc = (key >= SS);
#pragma unroll
      for (int r = 0; r < 8; ++r) {
        float sv = maskc ? -1e30f : sf[r] * 0.125f;   // 1/sqrt(64)
        float t = sv;
#pragma unroll
        for (int off = 1; off < 16; off <<= 1)
          t = fmaxf(t, __shfl_xor(t, off, 32));
        float mnew = fmaxf(mrow[r], t);
        float corr = __expf(mrow[r] - mnew);
        float p = __expf(sv - mnew);
        float ps = p;
#pragma unroll
        for (int off = 1; off < 16; off <<= 1) ps += __shfl_xor(ps, off, 32);
        lrow[r] = lrow[r] * corr + ps;
        mrow[r] = mnew;
        acc[0][r] *= corr; acc[1][r] *= corr;
        acc[2][r] *= corr; acc[3][r] *= corr;
        // C-layout -> LDS row-major P tile (row = r + 8*laneHi, col = key idx)
        lds_p[(r + 8 * laneHi) * 32 + kt * 16 + laneLo] = (f16)p;
      }
    }
    __syncthreads();
    v16h pa = load_a16(lds_p + laneLo * 32, laneHi);   // P as A-fragment 16x32
#pragma unroll
    for (int jt = 0; jt < 4; ++jt) {
      const f16* vp =
          vtws + ((size_t)bh * DH + jt * 16 + laneLo) * SP + kc + laneHi * 16;
      v16h vb = *(const v16h*)vp;                       // V^T: contiguous keys
      acc[jt] = wmma_f16(pa, vb, acc[jt]);
    }
    __syncthreads();
  }

#pragma unroll
  for (int r = 0; r < 8; ++r) {
    int s = qt * 16 + r + 8 * laneHi;
    if (s < SS) {
      float inv = 1.0f / lrow[r];
#pragma unroll
      for (int jt = 0; jt < 4; ++jt)
        out[((size_t)b * SS + s) * DD + h * DH + jt * 16 + laneLo] =
            acc[jt][r] * inv;
    }
  }
}

// ---------- launcher ----------

extern "C" void kernel_launch(void* const* d_in, const int* in_sizes, int n_in,
                              void* d_out, int out_size, void* d_ws,
                              size_t ws_size, hipStream_t stream) {
  (void)in_sizes; (void)n_in; (void)out_size; (void)ws_size;
  const float* hs = (const float*)d_in[0];
  const float* Wq = (const float*)d_in[1];
  const float* bq = (const float*)d_in[2];
  const float* Wk = (const float*)d_in[3];
  const float* bk = (const float*)d_in[4];
  const float* Wv = (const float*)d_in[5];
  const float* bv = (const float*)d_in[6];
  float* out = (float*)d_out;

  char* ws = (char*)d_ws;
  size_t off = 0;
  auto carve = [&](size_t bytes) -> void* {
    void* p = ws + off;
    off = (off + bytes + 255) & ~(size_t)255;
    return p;
  };
  const size_t xh_bytes  = (size_t)MROWS * DD * sizeof(f16);
  const size_t wt_bytes  = (size_t)DD * DD * sizeof(f16);
  const size_t qkv_bytes = (size_t)BB * HH * SP * DH * sizeof(f16);

  f16* xh   = (f16*)carve(xh_bytes);
  f16* wtq  = (f16*)carve(wt_bytes);
  f16* wtk  = (f16*)carve(wt_bytes);
  f16* wtv  = (f16*)carve(wt_bytes);
  f16* q_ws = (f16*)carve(qkv_bytes);   // [bh, SP, 64]
  f16* k_ws = (f16*)carve(qkv_bytes);   // [bh, SP, 64]
  f16* vt_ws= (f16*)carve(qkv_bytes);   // [bh, 64, SP]

  // 1) precision conversion + weight transpose
  f32_to_f16_kernel<<<4096, 256, 0, stream>>>(hs, xh, MROWS * DD);
  w_transpose_f16_kernel<<<2048, 256, 0, stream>>>(Wq, wtq);
  w_transpose_f16_kernel<<<2048, 256, 0, stream>>>(Wk, wtk);
  w_transpose_f16_kernel<<<2048, 256, 0, stream>>>(Wv, wtv);
  // zero K/V pads so padded keys contribute nothing (scores masked, P=0, V=0)
  hipMemsetAsync(k_ws, 0, qkv_bytes, stream);
  hipMemsetAsync(vt_ws, 0, qkv_bytes, stream);

  // 2) QKV projections (WMMA GEMM, fused bias + layout scatter)
  dim3 gg((MT32 + 7) / 8, DD / 64);
  qkv_gemm_kernel<<<gg, 256, 0, stream>>>(xh, wtq, bq, q_ws, 0);
  qkv_gemm_kernel<<<gg, 256, 0, stream>>>(xh, wtk, bk, k_ws, 1);
  qkv_gemm_kernel<<<gg, 256, 0, stream>>>(xh, wtv, bv, vt_ws, 2);

  // 3) attention (flash-style, WMMA scores + WMMA P*V)
  dim3 ga(BB * HH, (SS + 15) / 16);
  attn_kernel<<<ga, 32, 0, stream>>>(q_ws, k_ws, vt_ws, out);
}
